// RNNBlock_74028056314220
// MI455X (gfx1250) — compile-verified, matrix operand order checked by reading, not run on hardware
//
#include <hip/hip_runtime.h>

// ---------------------------------------------------------------------------
// RG-LRU bidirectional block for MI455X (gfx1250)
// bf16 WMMA GEMMs with async global->LDS double-buffered staging + chunked scan
// B=4, S=4096, D=Hd=1024
// ---------------------------------------------------------------------------

#define BB     4
#define SEQ    4096
#define HD     1024
#define MROWS  (BB * SEQ)      // 16384 token rows
#define CH     (BB * HD)       // 4096 scan channels
#define NCHUNK 64
#define CLEN   (SEQ / NCHUNK)  // 64

#define LDSPAD 72              // padded row stride (halves) = 144B, 16B-aligned

typedef __attribute__((ext_vector_type(16))) __bf16  v16bf;
typedef __attribute__((ext_vector_type(8)))  float   v8f;
typedef __attribute__((ext_vector_type(4)))  unsigned u32x4;

// ---- scalar helpers -------------------------------------------------------

__device__ __forceinline__ float bf2f(__bf16 b) {
  unsigned short hs = __builtin_bit_cast(unsigned short, b);
  unsigned u = ((unsigned)hs) << 16;
  return __builtin_bit_cast(float, u);
}

__device__ __forceinline__ __bf16 f2bf(float f) {
  unsigned u = __builtin_bit_cast(unsigned, f);
  u += 0x7fffu + ((u >> 16) & 1u);   // round-to-nearest-even
  unsigned short hs = (unsigned short)(u >> 16);
  return __builtin_bit_cast(__bf16, hs);
}

__device__ __forceinline__ float sigmoidf_(float x) {
  return 1.0f / (1.0f + __expf(-x));
}

__device__ __forceinline__ float gelu_tanh(float x) {
  float u = 0.7978845608028654f * (x + 0.044715f * x * x * x);
  return 0.5f * x * (1.0f + tanhf(u));
}

// ---- CDNA5 async copy primitives (ASYNCcnt-tracked DMA into LDS) ----------

__device__ __forceinline__ void async_cp16(unsigned lds_off, const void* gptr) {
  // VDST = LDS byte offset (hardware adds wave LDS_BASE), ADDR = 64-bit global
  asm volatile("global_load_async_to_lds_b128 %0, %1, off"
               :: "v"(lds_off), "v"(gptr)
               : "memory");
}

__device__ __forceinline__ void wait_async0() {
  asm volatile("s_wait_asynccnt 0" ::: "memory");
}

// ---- elementwise kernels --------------------------------------------------

__global__ void k_gelu_bf16(const float* __restrict__ x, __bf16* __restrict__ o, int n) {
  int i = blockIdx.x * blockDim.x + threadIdx.x;
  if (i < n) o[i] = f2bf(gelu_tanh(x[i]));
}

// fp32 [rows,cols] row-major  ->  bf16 [cols,rows] row-major (W^T)
__global__ void k_wconvT(const float* __restrict__ w, __bf16* __restrict__ wt,
                         int rows, int cols) {
  __shared__ float tile[32][33];
  int bx = blockIdx.x * 32, by = blockIdx.y * 32;
  for (int i = threadIdx.y; i < 32; i += 8)
    tile[i][threadIdx.x] = w[(size_t)(by + i) * cols + bx + threadIdx.x];
  __syncthreads();
  for (int i = threadIdx.y; i < 32; i += 8)
    wt[(size_t)(bx + i) * rows + by + threadIdx.x] = f2bf(tile[threadIdx.x][i]);
}

// xin = sqrt(1-a^2) * gate_x * x1
__global__ void k_xin(const float* __restrict__ a, const float* __restrict__ gx,
                      const __bf16* __restrict__ x1, float* __restrict__ xin, int n) {
  int i = blockIdx.x * blockDim.x + threadIdx.x;
  if (i < n) {
    float av = a[i];
    float m  = fmaxf(0.0f, 1.0f - av * av);
    xin[i] = sqrtf(m) * gx[i] * bf2f(x1[i]);
  }
}

// ---- chunked linear scan (h_t = a_t h_{t-1} + x_t), 3 passes --------------

__global__ void k_scan1(const float* __restrict__ a, const float* __restrict__ xin,
                        float* __restrict__ hl, float* __restrict__ pp,
                        float* __restrict__ carA, float* __restrict__ carH, int rev) {
  int t = blockIdx.x * blockDim.x + threadIdx.x;   // (chunk, channel)
  int c = t % CH, chunk = t / CH;
  int b = c / HD, hh = c % HD;
  float h = 0.0f, p = 1.0f;
  for (int i = 0; i < CLEN; ++i) {
    int u = chunk * CLEN + i;
    int s = rev ? (SEQ - 1 - u) : u;
    size_t idx = ((size_t)(b * SEQ + s)) * HD + hh;
    float aa = a[idx];
    h = fmaf(aa, h, xin[idx]);
    p *= aa;
    hl[idx] = h;
    pp[idx] = p;
  }
  carA[t] = p;
  carH[t] = h;
}

__global__ void k_scan2(const float* __restrict__ carA, const float* __restrict__ carH,
                        float* __restrict__ carIn) {
  int c = blockIdx.x * blockDim.x + threadIdx.x;   // channel
  float carry = 0.0f;
  for (int ch = 0; ch < NCHUNK; ++ch) {
    int idx = ch * CH + c;
    carIn[idx] = carry;
    carry = fmaf(carA[idx], carry, carH[idx]);
  }
}

__global__ void k_scan3(const float* __restrict__ hl, const float* __restrict__ pp,
                        const float* __restrict__ carIn, __bf16* __restrict__ hb, int rev) {
  int t = blockIdx.x * blockDim.x + threadIdx.x;   // (u, channel)
  int c = t % CH, u = t / CH;
  int b = c / HD, hh = c % HD;
  int chunk = u / CLEN;
  int s = rev ? (SEQ - 1 - u) : u;
  size_t idx = ((size_t)(b * SEQ + s)) * HD + hh;
  float h = fmaf(carIn[chunk * CH + c], pp[idx], hl[idx]);
  hb[idx] = f2bf(h);
}

// ---- WMMA bf16 GEMM, LDS-staged via async copies --------------------------
// C[M,N] = A[M,K] * Wt[N,K]^T + bias, fused epilogues.
// Block: 256 thr = 8 waves (4 M x 2 N), tile 128x64, BK=64, double buffered.

__device__ __forceinline__ v16bf lds_frag(const __bf16* __restrict__ tile,
                                          int r0, int kl, int lane) {
  // 16-bit A/B fragment lane layout (ISA 7.12.2):
  //   lanes 0-15 : K = kl+0..7  and kl+16..23   (row = r0 + lane)
  //   lanes 16-31: K = kl+8..15 and kl+24..31   (row = r0 + lane-16)
  int m  = lane & 15;
  int kb = (lane >> 4) << 3;
  const __bf16* q = tile + (r0 + m) * LDSPAD + kl + kb;
  union { u32x4 u[2]; v16bf v; } fr;
  fr.u[0] = *reinterpret_cast<const u32x4*>(q);
  fr.u[1] = *reinterpret_cast<const u32x4*>(q + 16);
  return fr.v;
}

// Issue async DMA for one K-stage: A 128x64 halves, B 64x64 halves (16B chunks)
__device__ __forceinline__ void stage_load(const __bf16* __restrict__ A,
                                           const __bf16* __restrict__ Wt,
                                           unsigned ldsA, unsigned ldsB,
                                           int rowA0, int rowB0, int K, int k0,
                                           int tid) {
#pragma unroll
  for (int i = 0; i < 4; ++i) {          // A: 1024 chunks / 256 threads
    int q = tid + 256 * i;
    int r = q >> 3, c = (q & 7) * 8;     // row, half-column
    async_cp16(ldsA + (unsigned)(r * LDSPAD + c) * 2,
               A + (size_t)(rowA0 + r) * K + k0 + c);
  }
#pragma unroll
  for (int i = 0; i < 2; ++i) {          // B: 512 chunks / 256 threads
    int q = tid + 256 * i;
    int r = q >> 3, c = (q & 7) * 8;
    async_cp16(ldsB + (unsigned)(r * LDSPAD + c) * 2,
               Wt + (size_t)(rowB0 + r) * K + k0 + c);
  }
}

// MODE 0: out0(bf16)  = C + bias                                (x1)
// MODE 1: out0(f32)   = sigmoid(C + bias)                       (gate_x)
// MODE 2: out0(f32)   = sigmoid(C + bias) * sigmoid(ex0[col])   (a = gate_a*a_expit)
// MODE 3: out0(f32)   = C + bias                                (fwd h@Wo + bo)
// MODE 4: t = (ex0 + C + bias)*0.5 + ex1; out0(f32)=t; out1(bf16)=gelu(t)
// MODE 5: out0(f32)   = C + bias + ex0                          (last dense + residual)
template <int MODE>
__global__ __launch_bounds__(256, 2)
void k_gemm(const __bf16* __restrict__ A, const __bf16* __restrict__ Wt,
            const float* __restrict__ bias,
            void* __restrict__ out0, void* __restrict__ out1,
            const float* __restrict__ ex0, const float* __restrict__ ex1,
            int N, int K) {
  __shared__ __bf16 As[2][128][LDSPAD];
  __shared__ __bf16 Bs[2][64][LDSPAD];

  const int tid  = threadIdx.x;
  const int lane = tid & 31;
  const int wid  = tid >> 5;              // 8 waves: 4 in M x 2 in N
  const int rowW = (wid & 3) * 32;        // wave tile origin (local)
  const int colW = (wid >> 2) * 32;
  const int rowA0 = blockIdx.x * 128;
  const int rowB0 = blockIdx.y * 64;

  const unsigned ldsA0 = (unsigned)(size_t)(void*)&As[0][0][0];
  const unsigned ldsA1 = (unsigned)(size_t)(void*)&As[1][0][0];
  const unsigned ldsB0 = (unsigned)(size_t)(void*)&Bs[0][0][0];
  const unsigned ldsB1 = (unsigned)(size_t)(void*)&Bs[1][0][0];

  v8f acc00 = {}, acc01 = {}, acc10 = {}, acc11 = {};

  const int nst = K / 64;
  stage_load(A, Wt, ldsA0, ldsB0, rowA0, rowB0, K, 0, tid);

  int buf = 0;
  for (int s = 0; s < nst; ++s) {
    wait_async0();        // my DMA done
    __syncthreads();      // everyone's DMA published; prior-stage reads done
    if (s + 1 < nst)
      stage_load(A, Wt, buf ? ldsA0 : ldsA1, buf ? ldsB0 : ldsB1,
                 rowA0, rowB0, K, (s + 1) * 64, tid);

    const __bf16* At = &As[buf][0][0];
    const __bf16* Bt = &Bs[buf][0][0];
#pragma unroll
    for (int kl = 0; kl < 64; kl += 32) {
      v16bf a0 = lds_frag(At, rowW,      kl, lane);
      v16bf a1 = lds_frag(At, rowW + 16, kl, lane);
      v16bf b0 = lds_frag(Bt, colW,      kl, lane);
      v16bf b1 = lds_frag(Bt, colW + 16, kl, lane);
      acc00 = __builtin_amdgcn_wmma_f32_16x16x32_bf16(false, a0, false, b0, (short)0, acc00, false, false);
      acc01 = __builtin_amdgcn_wmma_f32_16x16x32_bf16(false, a0, false, b1, (short)0, acc01, false, false);
      acc10 = __builtin_amdgcn_wmma_f32_16x16x32_bf16(false, a1, false, b0, (short)0, acc10, false, false);
      acc11 = __builtin_amdgcn_wmma_f32_16x16x32_bf16(false, a1, false, b1, (short)0, acc11, false, false);
    }
    buf ^= 1;
  }

  // C/D layout (ISA 7.12.2): element v of v8f at row (v + 8*(lane>=16)), col lane&15
  const int mlo = (lane >> 4) << 3;
  const int nn  = lane & 15;
  v8f accs[4] = {acc00, acc01, acc10, acc11};
#pragma unroll
  for (int f = 0; f < 4; ++f) {
    int fm = f >> 1, fn = f & 1;
    int col = blockIdx.y * 64 + colW + fn * 16 + nn;
    float bv = bias[col];
#pragma unroll
    for (int v = 0; v < 8; ++v) {
      int row = rowA0 + rowW + fm * 16 + mlo + v;
      size_t idx = (size_t)row * N + col;
      float c = accs[f][v] + bv;
      if (MODE == 0) {
        ((__bf16*)out0)[idx] = f2bf(c);
      } else if (MODE == 1) {
        ((float*)out0)[idx] = sigmoidf_(c);
      } else if (MODE == 2) {
        ((float*)out0)[idx] = sigmoidf_(c) * sigmoidf_(ex0[col]);
      } else if (MODE == 3) {
        ((float*)out0)[idx] = c;
      } else if (MODE == 4) {
        float t = (ex0[idx] + c) * 0.5f + ex1[idx];
        ((float*)out0)[idx]  = t;
        ((__bf16*)out1)[idx] = f2bf(gelu_tanh(t));
      } else {
        ((float*)out0)[idx] = c + ex0[idx];
      }
    }
  }
}

// ---------------------------------------------------------------------------

extern "C" void kernel_launch(void* const* d_in, const int* in_sizes, int n_in,
                              void* d_out, int out_size, void* d_ws, size_t ws_size,
                              hipStream_t stream) {
  (void)in_sizes; (void)n_in; (void)out_size; (void)ws_size;

  const float* x       = (const float*)d_in[0];
  const float* f_alog  = (const float*)d_in[1];
  const float* f_Wx    = (const float*)d_in[2];
  const float* f_bx    = (const float*)d_in[3];
  const float* f_Wgx   = (const float*)d_in[4];
  const float* f_bgx   = (const float*)d_in[5];
  const float* f_Wga   = (const float*)d_in[6];
  const float* f_bga   = (const float*)d_in[7];
  const float* f_Wo    = (const float*)d_in[8];
  const float* f_bo    = (const float*)d_in[9];
  const float* b_alog  = (const float*)d_in[10];
  const float* b_Wx    = (const float*)d_in[11];
  const float* b_bx    = (const float*)d_in[12];
  const float* b_Wgx   = (const float*)d_in[13];
  const float* b_bgx   = (const float*)d_in[14];
  const float* b_Wga   = (const float*)d_in[15];
  const float* b_bga   = (const float*)d_in[16];
  const float* b_Wo    = (const float*)d_in[17];
  const float* b_bo    = (const float*)d_in[18];
  const float* ld_W    = (const float*)d_in[19];
  const float* ld_b    = (const float*)d_in[20];
  float* out           = (float*)d_out;

  const size_t NE = (size_t)MROWS * HD;   // 16M elements

  // workspace carve-up
  char* ws = (char*)d_ws;
  size_t off = 0;
  auto carve = [&](size_t bytes) -> void* {
    void* p = ws + off;
    off += (bytes + 255) & ~(size_t)255;
    return p;
  };
  __bf16* xg   = (__bf16*)carve(NE * 2);
  __bf16* wt[9];
  for (int i = 0; i < 9; ++i) wt[i] = (__bf16*)carve((size_t)1024 * 1024 * 2);
  __bf16* x1b  = (__bf16*)carve(NE * 2);
  float*  gx   = (float*) carve(NE * 4);
  float*  av   = (float*) carve(NE * 4);
  float*  xin  = (float*) carve(NE * 4);
  float*  hl   = (float*) carve(NE * 4);
  float*  pp   = (float*) carve(NE * 4);
  __bf16* hb   = (__bf16*)carve(NE * 2);
  float*  accb = (float*) carve(NE * 4);
  float*  tbuf = (float*) carve(NE * 4);
  __bf16* gb   = (__bf16*)carve(NE * 2);
  float*  carA = (float*) carve((size_t)CH * NCHUNK * 4);
  float*  carH = (float*) carve((size_t)CH * NCHUNK * 4);
  float*  carI = (float*) carve((size_t)CH * NCHUNK * 4);

  const dim3 gemm_grid(MROWS / 128, HD / 64);
  const dim3 gemm_blk(256);
  const int  EWB = 256;
  const int  n_blocks = (int)(NE / EWB);

  // 1) xg = bf16(gelu(x))
  k_gelu_bf16<<<n_blocks, EWB, 0, stream>>>(x, xg, (int)NE);

  // 2) convert + transpose all 9 weight matrices to bf16 [N,K]
  const float* wsrc[9] = {f_Wx, f_Wgx, f_Wga, f_Wo, b_Wx, b_Wgx, b_Wga, b_Wo, ld_W};
  for (int i = 0; i < 9; ++i)
    k_wconvT<<<dim3(32, 32), dim3(32, 8), 0, stream>>>(wsrc[i], wt[i], 1024, 1024);

  // per-direction pipeline
  auto run_dir = [&](const float* alog, __bf16* wWx, __bf16* wWgx, __bf16* wWga,
                     __bf16* wWo, const float* bx, const float* bgx,
                     const float* bga, const float* bo, int rev) {
    // x1 = xg @ Wx + bx  (bf16)
    k_gemm<0><<<gemm_grid, gemm_blk, 0, stream>>>(xg, wWx, bx, x1b, nullptr, nullptr, nullptr, HD, HD);
    // gate_x = sigmoid(x1 @ Wgx + bgx)
    k_gemm<1><<<gemm_grid, gemm_blk, 0, stream>>>(x1b, wWgx, bgx, gx, nullptr, nullptr, nullptr, HD, HD);
    // a = sigmoid(x1 @ Wga + bga) * sigmoid(a_logit)
    k_gemm<2><<<gemm_grid, gemm_blk, 0, stream>>>(x1b, wWga, bga, av, nullptr, alog, nullptr, HD, HD);
    // xin = sqrt(1-a^2) * gate_x * x1
    k_xin<<<n_blocks, EWB, 0, stream>>>(av, gx, x1b, xin, (int)NE);
    // chunked linear scan -> hb (bf16)
    k_scan1<<<(CH * NCHUNK) / EWB, EWB, 0, stream>>>(av, xin, hl, pp, carA, carH, rev);
    k_scan2<<<CH / EWB, EWB, 0, stream>>>(carA, carH, carI);
    k_scan3<<<n_blocks, EWB, 0, stream>>>(hl, pp, carI, hb, rev);
    if (rev == 0) {
      // accb = h @ Wo + bo   (fwd)
      k_gemm<3><<<gemm_grid, gemm_blk, 0, stream>>>(hb, wWo, bo, accb, nullptr, nullptr, nullptr, HD, HD);
    } else {
      // t = (accb + h@Wo + bo)/2 + x ; gb = bf16(gelu(t))
      k_gemm<4><<<gemm_grid, gemm_blk, 0, stream>>>(hb, wWo, bo, tbuf, gb, accb, x, HD, HD);
    }
  };

  run_dir(f_alog, wt[0], wt[1], wt[2], wt[3], f_bx, f_bgx, f_bga, f_bo, 0);
  run_dir(b_alog, wt[4], wt[5], wt[6], wt[7], b_bx, b_bgx, b_bga, b_bo, 1);

  // out = gelu(t) @ ld_W + ld_b + t
  k_gemm<5><<<gemm_grid, gemm_blk, 0, stream>>>(gb, wt[8], ld_b, out, nullptr, tbuf, nullptr, HD, HD);
}
